// OrthoInitPhasor_48232482734973
// MI455X (gfx1250) — matrix operand order repaired and checked by reading, not compiled, not run on hardware
//
#include <hip/hip_runtime.h>
#include <hip/hip_bf16.h>
#include <math.h>

#define DIM   1024
#define SEQ   4096
#define BATCH 4
#define CHUNK 64
#define MTOT  (BATCH * SEQ)   // 16384 rows

#define LDSP  40              // LDS row pitch in ushorts (80B = 20 dwords -> conflict-free frags)

typedef __attribute__((ext_vector_type(16))) __bf16         v16bf;
typedef __attribute__((ext_vector_type(8)))  float          v8f;
typedef __attribute__((ext_vector_type(8)))  unsigned short ushort8;

// ---------- forced-global (addrspace 1) access helpers ----------
// Loop-carried pointer PHIs can defeat the addrspace inference pass and produce
// flat_load (which also ties up DScnt and wrecks the WMMA ds-wait pipelining).
// Global apertures are identity-mapped in flat space, so an integer round-trip
// cast to AS(1) is safe and guarantees global_load/global_store selection.
__device__ __forceinline__ ushort8 gload_u16x8(const unsigned short* p) {
    return *(const __attribute__((address_space(1))) ushort8*)(unsigned long long)p;
}
__device__ __forceinline__ float gload_f32(const float* p) {
    return *(const __attribute__((address_space(1))) float*)(unsigned long long)p;
}
__device__ __forceinline__ void gstore_f32(float* p, float v) {
    *(__attribute__((address_space(1))) float*)(unsigned long long)p = v;
}
__device__ __forceinline__ void gstore_u16(unsigned short* p, unsigned short v) {
    *(__attribute__((address_space(1))) unsigned short*)(unsigned long long)p = v;
}

// ---------- bf16 split helpers (round-to-nearest-even) ----------
__device__ __forceinline__ unsigned short f2bf_rne(float f) {
    unsigned int u = __float_as_uint(f);
    unsigned int r = u + 0x7FFFu + ((u >> 16) & 1u);
    return (unsigned short)(r >> 16);
}
__device__ __forceinline__ float bfbits2f(unsigned short h) {
    return __uint_as_float(((unsigned int)h) << 16);
}
__device__ __forceinline__ void split2(float x, unsigned short* hi, unsigned short* lo) {
    unsigned short h = f2bf_rne(x);
    gstore_u16(hi, h);
    gstore_u16(lo, f2bf_rne(x - bfbits2f(h)));
}

// ---------- split fp32 -> (bf16 hi, bf16 lo) planes ----------
__global__ void split_kernel(const float* __restrict__ in,
                             unsigned short* __restrict__ hi,
                             unsigned short* __restrict__ lo, int n) {
    int i = blockIdx.x * blockDim.x + threadIdx.x;
    int stride = gridDim.x * blockDim.x;
    for (; i < n; i += stride) split2(in[i], &hi[i], &lo[i]);
}

// ---------- transpose + split: W[r][c] fp32 -> T[c][r] bf16 hi/lo planes ----------
__global__ __launch_bounds__(256) void transpose_split_kernel(
    const float* __restrict__ W,
    unsigned short* __restrict__ Th, unsigned short* __restrict__ Tl)
{
    __shared__ float tile[32][33];
    const int tx = threadIdx.x & 31;
    const int ty = threadIdx.x >> 5;     // 0..7
    const int bx = blockIdx.x;           // source col tile
    const int by = blockIdx.y;           // source row tile
#pragma unroll
    for (int i = 0; i < 4; ++i)
        tile[ty + i * 8][tx] = W[(size_t)(by * 32 + ty + i * 8) * DIM + bx * 32 + tx];
    __syncthreads();
#pragma unroll
    for (int i = 0; i < 4; ++i) {
        const int orow = bx * 32 + ty + i * 8;   // = source col
        const int ocol = by * 32 + tx;           // = source row
        const float v = tile[tx][ty + i * 8];    // = W[ocol][orow]
        split2(v, &Th[(size_t)orow * DIM + ocol], &Tl[(size_t)orow * DIM + ocol]);
    }
}

// ---------- bf16x3 GEMM: C = A*B + bias (+resid) ----------
// A[M,1024] hi/lo planes (row-major), B given PRE-TRANSPOSED: Bt[N=1024][K=1024] hi/lo planes.
// Accumulates Ah*Bh + Ah*Bl + Al*Bh in fp32 (bf16x3 ~ fp32 accuracy).
// Block tile 128x128, 256 threads = 8 wave32s, each wave 32x64 (2x4 of 16x16 WMMA).
// Register-pipelined: step s+1 global loads issued before step s compute.
__global__ __launch_bounds__(256) void gemm_bf16x3(
    const unsigned short* __restrict__ Ah,  const unsigned short* __restrict__ Al,
    const unsigned short* __restrict__ Bth, const unsigned short* __restrict__ Btl,
    const float* __restrict__ bias, const float* __restrict__ resid,
    float* __restrict__ Cf, unsigned short* __restrict__ Chi, unsigned short* __restrict__ Clo)
{
    __shared__ unsigned short As [128][LDSP];   // A tile  [m][k]
    __shared__ unsigned short Bst[128][LDSP];   // B tile  [n][k] (from Bt, no transpose needed)

    const int t    = threadIdx.x;
    const int wid  = t >> 5;
    const int lane = t & 31;
    const int half = lane >> 4;
    const int lr   = lane & 15;
    const int wm   = wid >> 1;   // 0..3 (32-row stripes)
    const int wn   = wid & 1;    // 0..1 (64-col stripes)
    const int r0   = blockIdx.y * 128;
    const int c0   = blockIdx.x * 128;

    v8f acc[2][4];
#pragma unroll
    for (int i = 0; i < 2; ++i)
#pragma unroll
        for (int j = 0; j < 4; ++j)
            acc[i][j] = (v8f){0.f, 0.f, 0.f, 0.f, 0.f, 0.f, 0.f, 0.f};

    const int ldr = t >> 2;          // 0..63 : tile row this thread stages
    const int lds = (t & 3) * 8;     // 0/8/16/24 : k-segment

    const unsigned short* const Aps[3] = {Ah,  Ah,  Al };
    const unsigned short* const Bps[3] = {Bth, Btl, Bth};

    const int NSTEP = 3 * (DIM / 32);   // 96
    ushort8 ra0, ra1, rb0, rb1;
    {   // prologue: load step 0
        ra0 = gload_u16x8(Aps[0] + (size_t)(r0 + ldr)      * DIM + lds);
        ra1 = gload_u16x8(Aps[0] + (size_t)(r0 + ldr + 64) * DIM + lds);
        rb0 = gload_u16x8(Bps[0] + (size_t)(c0 + ldr)      * DIM + lds);
        rb1 = gload_u16x8(Bps[0] + (size_t)(c0 + ldr + 64) * DIM + lds);
    }

    for (int s = 0; s < NSTEP; ++s) {
        __syncthreads();   // previous compute done reading LDS
        *(ushort8*)(&As [ldr     ][lds]) = ra0;
        *(ushort8*)(&As [ldr + 64][lds]) = ra1;
        *(ushort8*)(&Bst[ldr     ][lds]) = rb0;
        *(ushort8*)(&Bst[ldr + 64][lds]) = rb1;
        __syncthreads();   // tile visible

        if (s + 1 < NSTEP) {   // issue next step's loads; consumed next iteration
            const int s1   = s + 1;
            const int pass = s1 >> 5;
            const int kt   = (s1 & 31) << 5;
            const unsigned short* Ap = Aps[pass];
            const unsigned short* Bp = Bps[pass];
            ra0 = gload_u16x8(Ap + (size_t)(r0 + ldr)      * DIM + kt + lds);
            ra1 = gload_u16x8(Ap + (size_t)(r0 + ldr + 64) * DIM + kt + lds);
            rb0 = gload_u16x8(Bp + (size_t)(c0 + ldr)      * DIM + kt + lds);
            rb1 = gload_u16x8(Bp + (size_t)(c0 + ldr + 64) * DIM + kt + lds);
        }
        if (s + 2 < NSTEP) {   // warm L2 two steps ahead
            const int s2   = s + 2;
            const int pass = s2 >> 5;
            const int kt   = (s2 & 31) << 5;
            __builtin_prefetch(Aps[pass] + (size_t)(r0 + ldr) * DIM + kt + lds, 0, 1);
            __builtin_prefetch(Bps[pass] + (size_t)(c0 + ldr) * DIM + kt + lds, 0, 1);
        }

        // Fragments per ISA 16-bit A/B layouts (CDNA5 ISA 7.12.2); all 16B LDS reads,
        // pitch 80B => each of 16 lanes maps to a distinct bank-quad (conflict-free).
        union { ushort8 u[2]; v16bf v; } af[2], bfr[4];
#pragma unroll
        for (int mi = 0; mi < 2; ++mi) {
            const unsigned short* ar = &As[wm * 32 + mi * 16 + lr][0];
            af[mi].u[0] = *(const ushort8*)(ar + half * 8);        // K = half*8 .. +7
            af[mi].u[1] = *(const ushort8*)(ar + 16 + half * 8);   // K = 16+half*8 .. +7
        }
#pragma unroll
        for (int ni = 0; ni < 4; ++ni) {
            const unsigned short* br = &Bst[wn * 64 + ni * 16 + lr][0];
            bfr[ni].u[0] = *(const ushort8*)(br + half * 8);       // K = half*8 .. +7
            bfr[ni].u[1] = *(const ushort8*)(br + 16 + half * 8);  // K = 16+half*8 .. +7
        }
#pragma unroll
        for (int mi = 0; mi < 2; ++mi)
#pragma unroll
            for (int ni = 0; ni < 4; ++ni)
                acc[mi][ni] = __builtin_amdgcn_wmma_f32_16x16x32_bf16(
                    false, af[mi].v, false, bfr[ni].v,
                    (short)0, acc[mi][ni], false, false);
    }

    // epilogue: bias, optional residual, fp32 out and/or bf16 hi/lo re-split
#pragma unroll
    for (int mi = 0; mi < 2; ++mi) {
#pragma unroll
        for (int ni = 0; ni < 4; ++ni) {
            const int col = c0 + wn * 64 + ni * 16 + lr;
            const float bv = bias ? bias[col] : 0.0f;
#pragma unroll
            for (int v = 0; v < 8; ++v) {
                const int row = r0 + wm * 32 + mi * 16 + half * 8 + v;  // C: VGPR v -> M = v + 8*half
                const size_t idx = (size_t)row * DIM + col;
                float val = acc[mi][ni][v] + bv;
                if (resid) val += gload_f32(&resid[idx]);
                if (Cf)  gstore_f32(&Cf[idx], val);
                if (Chi) split2(val, &Chi[idx], &Clo[idx]);
            }
        }
    }
}

// ---------- phasor binding + intra-chunk cumsum + retrieval ----------
__global__ __launch_bounds__(256) void phasor_kernel(
    const float* __restrict__ V, const float* __restrict__ KP, const float* __restrict__ QP,
    const float* __restrict__ base, const float* __restrict__ msptr,
    float* __restrict__ ret)
{
    const float ms = msptr[0];
    const int b  = blockIdx.x >> 6;   // SEQ/CHUNK = 64 chunks per batch
    const int c  = blockIdx.x & 63;
    const int s0 = c * CHUNK;
    const int t  = threadIdx.x;
    const float scale = 0.03125f;     // 1/sqrt(1024)
    float mr[4] = {0.f, 0.f, 0.f, 0.f};
    float mi[4] = {0.f, 0.f, 0.f, 0.f};
    for (int r = 0; r < CHUNK; ++r) {
        const int s = s0 + r;
        const size_t rowi  = ((size_t)b * SEQ + s) * DIM;
        const size_t browi = (size_t)s * DIM;
#pragma unroll
        for (int j = 0; j < 4; ++j) {
            const int col = t + j * 256;
            const float bp = base[browi + col];
            const float v  = V[rowi + col];
            const float kp = bp + KP[rowi + col] * ms;
            const float qp = bp + QP[rowi + col] * ms;
            mr[j] = fmaf(v, __cosf(kp), mr[j]);
            mi[j] = fmaf(v, __sinf(kp), mi[j]);
            gstore_f32(&ret[rowi + col], (mr[j] * __cosf(qp) + mi[j] * __sinf(qp)) * scale);
        }
    }
}

// ---------- LayerNorm over D, output bf16 hi/lo planes ----------
__global__ __launch_bounds__(256) void ln_kernel(
    const float* __restrict__ X, const float* __restrict__ g, const float* __restrict__ bb,
    unsigned short* __restrict__ Nh, unsigned short* __restrict__ Nl)
{
    __shared__ float s1[256], s2[256];
    const int row = blockIdx.x, t = threadIdx.x;
    const size_t base = (size_t)row * DIM;
    float x[4]; float sum = 0.f, sq = 0.f;
#pragma unroll
    for (int j = 0; j < 4; ++j) {
        x[j] = X[base + t + j * 256];
        sum += x[j]; sq += x[j] * x[j];
    }
    s1[t] = sum; s2[t] = sq;
    __syncthreads();
    for (int off = 128; off > 0; off >>= 1) {
        if (t < off) { s1[t] += s1[t + off]; s2[t] += s2[t + off]; }
        __syncthreads();
    }
    const float mu   = s1[0] * (1.0f / DIM);
    const float var  = s2[0] * (1.0f / DIM) - mu * mu;
    const float rstd = rsqrtf(var + 1e-5f);
#pragma unroll
    for (int j = 0; j < 4; ++j) {
        const int col = t + j * 256;
        float n = (x[j] - mu) * rstd * g[col] + bb[col];
        split2(n, &Nh[base + col], &Nl[base + col]);
    }
}

extern "C" void kernel_launch(void* const* d_in, const int* in_sizes, int n_in,
                              void* d_out, int out_size, void* d_ws, size_t ws_size,
                              hipStream_t stream) {
    const float* x    = (const float*)d_in[0];
    const float* base = (const float*)d_in[1];
    const float* Wk   = (const float*)d_in[2];
    const float* bk   = (const float*)d_in[3];
    const float* Wv   = (const float*)d_in[4];
    const float* bv   = (const float*)d_in[5];
    const float* Wq   = (const float*)d_in[6];
    const float* bq   = (const float*)d_in[7];
    const float* Wkm  = (const float*)d_in[8];
    const float* bkm  = (const float*)d_in[9];
    const float* Wqm  = (const float*)d_in[10];
    const float* bqm  = (const float*)d_in[11];
    const float* ms   = (const float*)d_in[12];
    const float* lng  = (const float*)d_in[13];
    const float* lnb  = (const float*)d_in[14];
    const float* Wo   = (const float*)d_in[15];
    const float* bo   = (const float*)d_in[16];
    float* out = (float*)d_out;

    const size_t XN = (size_t)MTOT * DIM;   // 16,777,216 elems
    const size_t WN = (size_t)DIM * DIM;    //  1,048,576 elems

    // Workspace layout (peak ~344 MiB), with reuse:
    char* ws = (char*)d_ws;
    unsigned short* x_hi  = (unsigned short*)ws; ws += XN * 2;
    unsigned short* x_lo  = (unsigned short*)ws; ws += XN * 2;
    unsigned short* wpl   = (unsigned short*)ws; ws += 12 * WN * 2;  // 6 weights x (hi,lo), transposed
    unsigned short* kq_hi = (unsigned short*)ws; ws += XN * 2;
    unsigned short* kq_lo = (unsigned short*)ws; ws += XN * 2;
    float* KP = (float*)ws; ws += XN * 4;
    float* QP = (float*)ws; ws += XN * 4;
    float* Vv = (float*)ws; ws += XN * 4;
    float*          retr = (float*)x_hi;           // x planes dead after GEMMs 1-3
    unsigned short* n_hi = kq_hi;                  // kq planes dead after GEMM 5
    unsigned short* n_lo = kq_lo;

    unsigned short* wk_hi  = wpl + 0 * 2 * WN; unsigned short* wk_lo  = wk_hi  + WN;
    unsigned short* wv_hi  = wpl + 1 * 2 * WN; unsigned short* wv_lo  = wv_hi  + WN;
    unsigned short* wq_hi  = wpl + 2 * 2 * WN; unsigned short* wq_lo  = wq_hi  + WN;
    unsigned short* wkm_hi = wpl + 3 * 2 * WN; unsigned short* wkm_lo = wkm_hi + WN;
    unsigned short* wqm_hi = wpl + 4 * 2 * WN; unsigned short* wqm_lo = wqm_hi + WN;
    unsigned short* wo_hi  = wpl + 5 * 2 * WN; unsigned short* wo_lo  = wo_hi  + WN;

    // 1) split x; transpose+split weights (Bt planes live in L2 for all GEMM blocks)
    split_kernel<<<8192, 256, 0, stream>>>(x, x_hi, x_lo, (int)XN);
    dim3 tg(32, 32);
    transpose_split_kernel<<<tg, 256, 0, stream>>>(Wk,  wk_hi,  wk_lo);
    transpose_split_kernel<<<tg, 256, 0, stream>>>(Wv,  wv_hi,  wv_lo);
    transpose_split_kernel<<<tg, 256, 0, stream>>>(Wq,  wq_hi,  wq_lo);
    transpose_split_kernel<<<tg, 256, 0, stream>>>(Wkm, wkm_hi, wkm_lo);
    transpose_split_kernel<<<tg, 256, 0, stream>>>(Wqm, wqm_hi, wqm_lo);
    transpose_split_kernel<<<tg, 256, 0, stream>>>(Wo,  wo_hi,  wo_lo);

    dim3 gg(DIM / 128, MTOT / 128);  // (8, 128)

    // 2) key = x@Wk + bk -> bf16 planes;  KP = key@Wkm + bkm -> fp32
    gemm_bf16x3<<<gg, 256, 0, stream>>>(x_hi, x_lo, wk_hi, wk_lo, bk, nullptr,
                                        nullptr, kq_hi, kq_lo);
    gemm_bf16x3<<<gg, 256, 0, stream>>>(kq_hi, kq_lo, wkm_hi, wkm_lo, bkm, nullptr,
                                        KP, nullptr, nullptr);
    // 3) query = x@Wq + bq -> bf16 planes (reuse); QP = query@Wqm + bqm -> fp32
    gemm_bf16x3<<<gg, 256, 0, stream>>>(x_hi, x_lo, wq_hi, wq_lo, bq, nullptr,
                                        nullptr, kq_hi, kq_lo);
    gemm_bf16x3<<<gg, 256, 0, stream>>>(kq_hi, kq_lo, wqm_hi, wqm_lo, bqm, nullptr,
                                        QP, nullptr, nullptr);
    // 4) value = x@Wv + bv -> fp32
    gemm_bf16x3<<<gg, 256, 0, stream>>>(x_hi, x_lo, wv_hi, wv_lo, bv, nullptr,
                                        Vv, nullptr, nullptr);

    // 5) phasor bind + chunk cumsum + retrieve (overwrites dead x planes)
    phasor_kernel<<<BATCH * (SEQ / CHUNK), 256, 0, stream>>>(Vv, KP, QP, base, ms, retr);

    // 6) LayerNorm -> normed bf16 planes (reuse kq planes)
    ln_kernel<<<MTOT, 256, 0, stream>>>(retr, lng, lnb, n_hi, n_lo);

    // 7) out = x + normed@Wo + bo
    gemm_bf16x3<<<gg, 256, 0, stream>>>(n_hi, n_lo, wo_hi, wo_lo, bo, x,
                                        out, nullptr, nullptr);
}
